// VectorQuantizer2_31267361915590
// MI455X (gfx1250) — matrix operand order repaired and checked by reading, not compiled, not run on hardware
//
#include <hip/hip_runtime.h>

typedef __attribute__((ext_vector_type(16))) _Float16 v16h;
typedef __attribute__((ext_vector_type(8)))  float    v8f;

#define N_E        4096
#define E_DIM      32
#define HW         4096        // 64*64
#define ROWS       65536       // 16 * HW
#define OUT_ELEMS  2097152     // 16*32*64*64
#define NTILES     (N_E / 16)  // 256

// ---------------------------------------------------------------------------
// Prep: split codebook into f16 hi/lo planes (WMMA-ready row-major layout),
// per-code squared norms, and zero the loss accumulator slot.
// ---------------------------------------------------------------------------
__global__ void vq_prep_kernel(const float* __restrict__ emb,
                               float* __restrict__ enorm,
                               _Float16* __restrict__ e_hi,
                               _Float16* __restrict__ e_lo,
                               float* __restrict__ loss_slot) {
    int n = blockIdx.x * blockDim.x + threadIdx.x;
    if (n == 0) *loss_slot = 0.0f;
    if (n < N_E) {
        float s = 0.0f;
#pragma unroll
        for (int i = 0; i < E_DIM; ++i) {
            float v = emb[n * E_DIM + i];
            s += v * v;
            _Float16 h = (_Float16)v;
            e_hi[n * E_DIM + i] = h;
            e_lo[n * E_DIM + i] = (_Float16)(v - (float)h);
        }
        enorm[n] = s;
    }
}

// ---------------------------------------------------------------------------
// Main: one wave per 16-row M-tile. WMMA f16 split-precision distance GEMM,
// running argmin, codebook gather + straight-through output + loss.
// ---------------------------------------------------------------------------
__global__ __launch_bounds__(256)
void vq_main_kernel(const float* __restrict__ z,
                    const float* __restrict__ emb,
                    const float* __restrict__ enorm,
                    const _Float16* __restrict__ e_hi,
                    const _Float16* __restrict__ e_lo,
                    float* __restrict__ out,
                    float* __restrict__ loss_slot,
                    int* __restrict__ idx_out) {
    const int lane = threadIdx.x & 31;
    const int wave = threadIdx.x >> 5;
    const int s    = lane & 15;   // column / sub-lane id
    const int half = lane >> 4;   // K-half selector for A/B layouts

    const int mtile = blockIdx.x * 8 + wave;   // 512 blocks * 8 waves = 4096 tiles
    const int r0    = mtile * 16;              // first flat row of tile
    const int bb    = r0 >> 12;                // batch index (HW = 4096 rows per batch)
    const int hw0   = r0 & (HW - 1);           // h*64+w of first row (multiple of 16)

    // ---- Load A tile (16 z rows x K=32) and split fp32 -> f16 hi/lo --------
    // A 16x32 layout (contiguous k order chosen here, B matches): lane holds row M = lane&15.
    // element e (0..15): K = 16*half + e
    const float* zrow = z + bb * (E_DIM * HW) + (hw0 + s);  // [k] at zrow[k*HW]
    v16h a_hi, a_lo;
#pragma unroll
    for (int e = 0; e < 16; ++e) {
        const int k = 16 * half + e;
        float v = zrow[k * HW];
        _Float16 h = (_Float16)v;
        a_hi[e] = h;
        a_lo[e] = (_Float16)(v - (float)h);
    }

    float best[8];
    int   bidx[8];
#pragma unroll
    for (int c = 0; c < 8; ++c) { best[c] = 3.0e38f; bidx[c] = 0; }

    // Per-lane streaming pointers (B tiles are pre-split, WMMA-ready):
    // lane (s, half) reads e_hi/e_lo[n*32 + 16*half .. +15], n = nt*16 + s.
    const _Float16* ph = e_hi  + s * E_DIM + 16 * half;
    const _Float16* pl = e_lo  + s * E_DIM + 16 * half;
    const float*    pe = enorm + s;

    // One-tile compute step: 3 split-precision WMMAs + running argmin.
    auto step = [&](v16h b_hi, v16h b_lo, float en, int n) {
        v8f acc = {0.f, 0.f, 0.f, 0.f, 0.f, 0.f, 0.f, 0.f};
        acc = __builtin_amdgcn_wmma_f32_16x16x32_f16(false, a_hi, false, b_hi,
                                                     (short)0, acc, false, false);
        acc = __builtin_amdgcn_wmma_f32_16x16x32_f16(false, a_lo, false, b_hi,
                                                     (short)0, acc, false, false);
        acc = __builtin_amdgcn_wmma_f32_16x16x32_f16(false, a_hi, false, b_lo,
                                                     (short)0, acc, false, false);
        // C/D layout: VGPR c holds row M = c + 8*half for this lane's column.
#pragma unroll
        for (int c = 0; c < 8; ++c) {
            float sc = __builtin_fmaf(-2.0f, acc[c], en);
            if (sc < best[c]) {            // strict < : first min wins (argmin)
                best[c] = sc;              // v_cmp_lt + dual cndmask codegen
                bidx[c] = n;
            }
        }
    };

    // ---- Sweep all 256 code tiles, loads double-buffered one tile ahead ----
    v16h b_hi = *(const v16h*)ph;
    v16h b_lo = *(const v16h*)pl;
    float en  = *pe;

    for (int nt = 0; nt < NTILES - 1; ++nt) {
        ph += 16 * E_DIM;
        pl += 16 * E_DIM;
        pe += 16;
        v16h  nb_hi = *(const v16h*)ph;   // next tile: overlaps current compute
        v16h  nb_lo = *(const v16h*)pl;
        float nen   = *pe;

        step(b_hi, b_lo, en, nt * 16 + s);

        b_hi = nb_hi;
        b_lo = nb_lo;
        en   = nen;
    }
    step(b_hi, b_lo, en, (NTILES - 1) * 16 + s);   // peeled last tile

    // ---- Cross-lane argmin within each 16-lane column group ----------------
#pragma unroll
    for (int off = 8; off >= 1; off >>= 1) {
#pragma unroll
        for (int c = 0; c < 8; ++c) {
            float ov = __shfl_xor(best[c], off, 32);
            int   oi = __shfl_xor(bidx[c], off, 32);
            if (ov < best[c] || (ov == best[c] && oi < bidx[c])) {
                best[c] = ov; bidx[c] = oi;
            }
        }
    }
    int idxc[8];
#pragma unroll
    for (int c = 0; c < 8; ++c)
        idxc[c] = __shfl(bidx[c], 0, 16);  // broadcast within each 16-lane group

    if (s == 0) {
#pragma unroll
        for (int c = 0; c < 8; ++c)
            idx_out[r0 + half * 8 + c] = idxc[c];
    }

    // ---- Gather z_q, write BCHW output, accumulate loss --------------------
    // Half group `half` owns rows (half*8 .. half*8+7) => hw = hw0+half*8 .. +7.
    float lsum = 0.0f;
    const int base = bb * (E_DIM * HW) + hw0 + half * 8;
#pragma unroll
    for (int t = 0; t < 2; ++t) {
        const int d = s + 16 * t;                 // channel handled by this lane
        const float* zp = z   + base + d * HW;
        float*       op = out + base + d * HW;
        float4 z0 = *(const float4*)(zp + 0);
        float4 z1 = *(const float4*)(zp + 4);
        float q[8];
#pragma unroll
        for (int c = 0; c < 8; ++c)
            q[c] = emb[idxc[c] * E_DIM + d];
        float4 q0 = {q[0], q[1], q[2], q[3]};
        float4 q1 = {q[4], q[5], q[6], q[7]};
        *(float4*)(op + 0) = q0;
        *(float4*)(op + 4) = q1;
        float zz[8] = { z0.x, z0.y, z0.z, z0.w, z1.x, z1.y, z1.z, z1.w };
#pragma unroll
        for (int c = 0; c < 8; ++c) {
            float dlt = q[c] - zz[c];
            lsum += dlt * dlt;
        }
    }

    // loss = (1 + BETA) * mean((z_q - z)^2); wave-reduce then one atomic.
#pragma unroll
    for (int off = 16; off >= 1; off >>= 1)
        lsum += __shfl_xor(lsum, off, 32);
    if (lane == 0)
        atomicAdd(loss_slot, lsum * (1.25f / (float)OUT_ELEMS));
}

// ---------------------------------------------------------------------------
extern "C" void kernel_launch(void* const* d_in, const int* in_sizes, int n_in,
                              void* d_out, int out_size, void* d_ws, size_t ws_size,
                              hipStream_t stream) {
    const float* z   = (const float*)d_in[0];   // [16,32,64,64] f32
    const float* emb = (const float*)d_in[1];   // [4096,32] f32

    float* out       = (float*)d_out;           // z_q straight-through, BCHW
    float* loss_slot = out + OUT_ELEMS;         // scalar loss
    int*   idx_out   = (int*)d_out + OUT_ELEMS + 1;  // 65536 argmin indices

    // Workspace layout: enorm (16KB) | e_hi (256KB) | e_lo (256KB)
    float*    enorm = (float*)d_ws;
    _Float16* e_hi  = (_Float16*)((char*)d_ws + 16 * 1024);
    _Float16* e_lo  = (_Float16*)((char*)d_ws + 16 * 1024 + N_E * E_DIM * 2);

    vq_prep_kernel<<<N_E / 256, 256, 0, stream>>>(emb, enorm, e_hi, e_lo, loss_slot);
    vq_main_kernel<<<ROWS / 16 / 8, 256, 0, stream>>>(z, emb, enorm, e_hi, e_lo,
                                                      out, loss_slot, idx_out);
}
